// GarmentDisplacementNet_50903952392812
// MI455X (gfx1250) — matrix operand order repaired and checked by reading, not compiled, not run on hardware
//
#include <hip/hip_runtime.h>
#include <hip/hip_bf16.h>

typedef __attribute__((ext_vector_type(16))) __bf16 v16bf;
typedef __attribute__((ext_vector_type(8)))  float  v8f;
typedef __attribute__((ext_vector_type(4)))  __bf16 bf16x4;   // 8B
typedef __attribute__((ext_vector_type(8)))  __bf16 bf16x8;   // 16B

#define VREAL 12000
#define VPAD  12032      /* 188 * 64 */
#define LSPI  16

// ---------------- stage x: [2,12000,479] f32 -> [2,VPAD,512] bf16 zero-padded ----------------
__global__ void stage_x(const float* __restrict__ x, __bf16* __restrict__ xbf) {
    long i = (long)blockIdx.x * blockDim.x + threadIdx.x;
    const long total = 2L * VPAD * 512;
    if (i >= total) return;
    int c = (int)(i & 511);
    int v = (int)((i >> 9) % VPAD);
    int b = (int)(i / ((long)VPAD * 512));
    float val = (v < VREAL && c < 479) ? x[((long)b * VREAL + v) * 479 + c] : 0.0f;
    xbf[i] = (__bf16)val;
}

// ---------------- weight stage: f32 [Ks,N] -> bf16 transposed [N,Kd] zero-padded ----------------
__global__ void padT_f32_to_bf16(const float* __restrict__ src, int Ks, int N,
                                 __bf16* __restrict__ dst, int Kd) {
    long i = (long)blockIdx.x * blockDim.x + threadIdx.x;
    long total = (long)N * Kd;
    if (i >= total) return;
    int n = (int)(i / Kd), k = (int)(i % Kd);
    float v = (k < Ks) ? src[(long)k * N + n] : 0.0f;
    dst[i] = (__bf16)v;
}

// ---------------- pipelined staging helpers ----------------
struct StageRegs { bf16x4 a[4]; bf16x8 w0, w1; };

__device__ __forceinline__ void load_dense(StageRegs& rg, const __bf16* A, const __bf16* Wt,
                                           long abase, long wbase, int K, int tid) {
#pragma unroll
    for (int q = 0; q < 4; ++q) {
        int cid = (q << 7) + tid;               // 0..511 -> 4 contiguous bf16 each
        int r = cid >> 3, c = (cid & 7) << 2;
        rg.a[q] = *(const bf16x4*)&A[abase + (long)r * K + c];
    }
    int n = tid >> 1, kh = (tid & 1) << 4;
    const __bf16* wp = &Wt[wbase + (long)n * K + kh];
    rg.w0 = *(const bf16x8*)(wp);
    rg.w1 = *(const bf16x8*)(wp + 8);
}

__device__ __forceinline__ void load_spiral(StageRegs& rg, const __bf16* src,
                                            const int (*sidx)[LSPI], long bbase, int logC, int k0,
                                            const __bf16* Wt, long wbase, int K, int tid) {
    const int C = 1 << logC;
#pragma unroll
    for (int q = 0; q < 4; ++q) {
        int cid = (q << 7) + tid;
        int r = cid >> 3, c = (cid & 7) << 2;
        int kk = k0 + c;
        int j  = kk >> logC;                    // chunk never crosses a neighbor boundary
        int cc = kk & (C - 1);
        int sv = sidx[r][j];                    // in [0, VREAL]; row VREAL is all-zero
        rg.a[q] = *(const bf16x4*)&src[bbase + (long)sv * C + cc];
    }
    int n = tid >> 1, kh = (tid & 1) << 4;
    const __bf16* wp = &Wt[wbase + (long)n * K + kh];
    rg.w0 = *(const bf16x8*)(wp);
    rg.w1 = *(const bf16x8*)(wp + 8);
}

__device__ __forceinline__ void store_stage(const StageRegs& rg,
                                            __bf16 (*lA)[32], __bf16 (*lB)[32], int tid) {
#pragma unroll
    for (int q = 0; q < 4; ++q) {
        int cid = (q << 7) + tid;
        int r = cid >> 3, c = (cid & 7) << 2;
        *(bf16x4*)&lA[r][c] = rg.a[q];
    }
    int n = tid >> 1, kh = (tid & 1) << 4;
    *(bf16x8*)&lB[n][kh]     = rg.w0;
    *(bf16x8*)&lB[n][kh + 8] = rg.w1;
}

__device__ __forceinline__ void compute_tile(__bf16 (*lA)[32], __bf16 (*lB)[32],
                                             v8f acc[4], int lane, int wave) {
    v16bf afrag;
    const int m   = lane & 15;
    const int kh  = (lane >> 4) << 3;
    const int kh2 = (lane >> 4) << 4;
    const __bf16* arow = lA[(wave << 4) + m];
#pragma unroll
    for (int e = 0; e < 8; ++e) { afrag[e] = arow[kh + e]; afrag[8 + e] = arow[16 + kh + e]; }
#pragma unroll
    for (int nq = 0; nq < 4; ++nq) {
        v16bf bfrag;
        const __bf16* brow = lB[(nq << 4) + (lane & 15)];
#pragma unroll
        for (int e = 0; e < 8; ++e) { bfrag[e] = brow[kh + e]; bfrag[8 + e] = brow[16 + kh + e]; }
        acc[nq] = __builtin_amdgcn_wmma_f32_16x16x32_bf16(false, afrag, false, bfrag,
                                                          (short)0, acc[nq], false, false);
    }
}

__device__ __forceinline__ void epilogue_store(const v8f acc[4], int b, int v0, int n0, int N,
                                               const float* bias, const __bf16* residual,
                                               int do_relu, int maskV, __bf16* out,
                                               int lane, int wave) {
    const int mrow = (wave << 4) + ((lane >> 4) << 3);
#pragma unroll
    for (int nq = 0; nq < 4; ++nq) {
        int n = n0 + (nq << 4) + (lane & 15);
        float bn = bias ? bias[n] : 0.0f;
#pragma unroll
        for (int r = 0; r < 8; ++r) {
            int v = v0 + mrow + r;
            float val = acc[nq][r] + bn;
            if (residual) val += (float)residual[((long)b * VPAD + v) * N + n];
            if (do_relu)  val = fmaxf(val, 0.0f);
            if (v >= maskV) val = 0.0f;
            out[((long)b * VPAD + v) * N + n] = (__bf16)val;
        }
    }
}

// ---------------- dense bf16 WMMA GEMM, 64x64 tile, double-buffered LDS ----------------
// A: [2, VPAD, K] bf16 (fully padded) ; Wt: [N, K] bf16 ; out/residual: [2, VPAD, N] bf16
__global__ void wmma_dense(const __bf16* __restrict__ A, int K,
                           const __bf16* __restrict__ Wt, int N,
                           const float* __restrict__ bias,
                           const __bf16* residual,
                           int do_relu, int maskV,
                           __bf16* out) {
    __shared__ __attribute__((aligned(16))) __bf16 lA[2][64][32];
    __shared__ __attribute__((aligned(16))) __bf16 lB[2][64][32];
    const int b    = blockIdx.z;
    const int v0   = blockIdx.y << 6;
    const int n0   = blockIdx.x << 6;
    const int tid  = threadIdx.x;
    const int lane = tid & 31, wave = tid >> 5;

    v8f acc[4];
#pragma unroll
    for (int t = 0; t < 4; ++t)
#pragma unroll
        for (int i = 0; i < 8; ++i) acc[t][i] = 0.0f;

    const long abase  = ((long)b * VPAD + v0) * K;
    const long wbase  = (long)n0 * K;
    const int  nsteps = K >> 5;

    StageRegs rg;
    load_dense(rg, A, Wt, abase, wbase, K, tid);
    store_stage(rg, lA[0], lB[0], tid);
    __syncthreads();

    for (int s = 0; s < nsteps; ++s) {
        const int  cur  = s & 1;
        const bool more = (s + 1) < nsteps;
        if (more) load_dense(rg, A, Wt, abase + (s + 1) * 32, wbase + (s + 1) * 32, K, tid);
        compute_tile(lA[cur], lB[cur], acc, lane, wave);       // overlaps in-flight loads
        if (more) store_stage(rg, lA[cur ^ 1], lB[cur ^ 1], tid);
        __syncthreads();
    }

    epilogue_store(acc, b, v0, n0, N, bias, residual, do_relu, maskV, out, lane, wave);
}

// ---------------- spiral-gather bf16 WMMA GEMM, 64x64 tile, double-buffered LDS ----------------
// src: [2, VPAD, C] bf16, C = 1<<logC ; idx: [VREAL+1, LSPI] ; Wt: [N, K], K = LSPI*C
__global__ void wmma_spiral(const __bf16* __restrict__ src, int logC,
                            const int* __restrict__ idx,
                            const __bf16* __restrict__ Wt, int K, int N,
                            const float* __restrict__ bias,
                            const __bf16* residual,
                            int do_relu, int maskV,
                            __bf16* out) {
    __shared__ __attribute__((aligned(16))) __bf16 lA[2][64][32];
    __shared__ __attribute__((aligned(16))) __bf16 lB[2][64][32];
    __shared__ int sidx[64][LSPI];
    const int b    = blockIdx.z;
    const int v0   = blockIdx.y << 6;
    const int n0   = blockIdx.x << 6;
    const int tid  = threadIdx.x;
    const int lane = tid & 31, wave = tid >> 5;

    // stage the 64x16 index tile once per block
#pragma unroll
    for (int q = 0; q < 8; ++q) {
        int e = (q << 7) + tid;                 // 0..1023
        int r = e >> 4, j = e & 15;
        int vi = v0 + r; if (vi > VREAL) vi = VREAL;
        sidx[r][j] = idx[vi * LSPI + j];
    }
    __syncthreads();

    v8f acc[4];
#pragma unroll
    for (int t = 0; t < 4; ++t)
#pragma unroll
        for (int i = 0; i < 8; ++i) acc[t][i] = 0.0f;

    const long bbase  = (long)b * VPAD << logC;     // b * VPAD * C
    const long wbase  = (long)n0 * K;
    const int  nsteps = K >> 5;

    StageRegs rg;
    load_spiral(rg, src, sidx, bbase, logC, 0, Wt, wbase, K, tid);
    store_stage(rg, lA[0], lB[0], tid);
    __syncthreads();

    for (int s = 0; s < nsteps; ++s) {
        const int  cur  = s & 1;
        const bool more = (s + 1) < nsteps;
        if (more) load_spiral(rg, src, sidx, bbase, logC, (s + 1) * 32,
                              Wt, wbase + (s + 1) * 32, K, tid);
        compute_tile(lA[cur], lB[cur], acc, lane, wave);
        if (more) store_stage(rg, lA[cur ^ 1], lB[cur ^ 1], tid);
        __syncthreads();
    }

    epilogue_store(acc, b, v0, n0, N, bias, residual, do_relu, maskV, out, lane, wave);
}

// ---------------- masked global max-pool over vertices ----------------
__global__ void init_gmax(float* g) {
    int i = blockIdx.x * blockDim.x + threadIdx.x;
    if (i < 256) g[i] = -3.0e38f;
}

__global__ void maxpool_fs(const __bf16* __restrict__ fs, float* __restrict__ gmax) {
    int b  = blockIdx.y;
    int c  = threadIdx.x;          // 128 channels
    int v0 = blockIdx.x * 256;
    float m = -3.0e38f;
    for (int v = v0; v < v0 + 256 && v < VREAL; ++v)
        m = fmaxf(m, (float)fs[((long)b * VPAD + v) * 128 + c]);
    float* addr = &gmax[b * 128 + c];
    if (m >= 0.0f) atomicMax((int*)addr, __float_as_int(m));
    else           atomicMin((unsigned int*)addr, __float_as_uint(m));
}

// ---------------- concat [pfs | fs | broadcast(gmax)*zp] -> h512 ----------------
__global__ void concat_h(const __bf16* __restrict__ pfs, const __bf16* __restrict__ fs,
                         const float* __restrict__ gmax, __bf16* __restrict__ h) {
    long i = (long)blockIdx.x * blockDim.x + threadIdx.x;
    const long total = 2L * VPAD * 512;
    if (i >= total) return;
    int c = (int)(i & 511);
    int v = (int)((i >> 9) % VPAD);
    int b = (int)(i / ((long)VPAD * 512));
    __bf16 val;
    if (c < 256)      val = pfs[((long)b * VPAD + v) * 256 + c];
    else if (c < 384) val = fs[((long)b * VPAD + v) * 128 + (c - 256)];
    else              val = (v < VREAL) ? (__bf16)gmax[b * 128 + (c - 384)] : (__bf16)0.0f;
    h[i] = val;
}

// ---------------- final tiny layer: [*,128] @ Wo3[128,3] + bo3 -> f32 out ----------------
__global__ void out_final(const __bf16* __restrict__ ho2, const float* __restrict__ Wo3,
                          const float* __restrict__ bo3, float* __restrict__ out) {
    int t = blockIdx.x * blockDim.x + threadIdx.x;
    if (t >= 2 * VREAL * 3) return;
    int n = t % 3;
    int v = (t / 3) % VREAL;
    int b = t / (3 * VREAL);
    float s = bo3[n];
    const __bf16* row = &ho2[((long)b * VPAD + v) * 128];
    for (int k = 0; k < 128; ++k) s += (float)row[k] * Wo3[k * 3 + n];
    out[t] = s;
}

extern "C" void kernel_launch(void* const* d_in, const int* in_sizes, int n_in,
                              void* d_out, int out_size, void* d_ws, size_t ws_size,
                              hipStream_t stream) {
    (void)in_sizes; (void)n_in; (void)out_size; (void)ws_size;
    const float* x    = (const float*)d_in[0];
    const int*   idx  = (const int*)d_in[1];
    const float* Wp   = (const float*)d_in[2];
    const float* r1W  = (const float*)d_in[3];
    const float* r1b  = (const float*)d_in[4];
    const float* Wmid = (const float*)d_in[5];
    const float* rsW  = (const float*)d_in[6];
    const float* rsb  = (const float*)d_in[7];
    const float* Wo1  = (const float*)d_in[8];
    const float* bo1  = (const float*)d_in[9];
    const float* Wo2  = (const float*)d_in[10];
    const float* bo2  = (const float*)d_in[11];
    const float* Wo3  = (const float*)d_in[12];
    const float* bo3  = (const float*)d_in[13];

    char*  ws  = (char*)d_ws;
    size_t off = 0;
    auto alloc = [&](size_t bytes) -> void* {
        void* p = ws + off;
        off = (off + bytes + 255) & ~(size_t)255;
        return p;
    };

    __bf16* xbf   = (__bf16*)alloc(2L * VPAD * 512 * 2);
    __bf16* Wpbf  = (__bf16*)alloc(256L * 512 * 2);
    __bf16* r1Wbf = (__bf16*)alloc(2L * 256 * 4096 * 2);
    __bf16* Wmbf  = (__bf16*)alloc(128L * 256 * 2);
    __bf16* rsWbf = (__bf16*)alloc(6L * 128 * 2048 * 2);
    __bf16* Wo1bf = (__bf16*)alloc(256L * 512 * 2);
    __bf16* Wo2bf = (__bf16*)alloc(128L * 256 * 2);
    __bf16* pfs   = (__bf16*)alloc(2L * VPAD * 256 * 2);
    __bf16* h256  = (__bf16*)alloc(2L * VPAD * 256 * 2);
    __bf16* r1    = (__bf16*)alloc(2L * VPAD * 256 * 2);   // later reused as ho1
    __bf16* fs    = (__bf16*)alloc(2L * VPAD * 128 * 2);
    __bf16* h128  = (__bf16*)alloc(2L * VPAD * 128 * 2);   // later reused as ho2
    __bf16* h512  = (__bf16*)alloc(2L * VPAD * 512 * 2);
    float*  gmax  = (float*)alloc(256 * 4);

    auto padgrid = [](long n) { return dim3((unsigned)((n + 255) / 256)); };

    // --- bf16 staging: x padded [2,VPAD,512]; weights transposed to [N,K] ---
    stage_x<<<padgrid(2L * VPAD * 512), 256, 0, stream>>>(x, xbf);
    padT_f32_to_bf16<<<padgrid(256L * 512), 256, 0, stream>>>(Wp, 479, 256, Wpbf, 512);
    for (int k = 0; k < 2; ++k)
        padT_f32_to_bf16<<<padgrid(256L * 4096), 256, 0, stream>>>(
            r1W + (long)k * 4096 * 256, 4096, 256, r1Wbf + (long)k * 256 * 4096, 4096);
    padT_f32_to_bf16<<<padgrid(128L * 256), 256, 0, stream>>>(Wmid, 256, 128, Wmbf, 256);
    for (int s = 0; s < 6; ++s)
        padT_f32_to_bf16<<<padgrid(128L * 2048), 256, 0, stream>>>(
            rsW + (long)s * 2048 * 128, 2048, 128, rsWbf + (long)s * 128 * 2048, 2048);
    padT_f32_to_bf16<<<padgrid(256L * 512), 256, 0, stream>>>(Wo1, 512, 256, Wo1bf, 512);
    padT_f32_to_bf16<<<padgrid(128L * 256), 256, 0, stream>>>(Wo2, 256, 128, Wo2bf, 256);

    dim3 blk(128);
    const dim3 g256(4, VPAD / 64, 2);   // N=256 tiles
    const dim3 g128(2, VPAD / 64, 2);   // N=128 tiles

    // pointMLP: pf = relu(x @ W_point), dummy+pad rows zeroed
    wmma_dense<<<g256, blk, 0, stream>>>(xbf, 512, Wpbf, 256,
                                         nullptr, nullptr, 1, VREAL, pfs);
    // res1 block (256ch)
    wmma_spiral<<<g256, blk, 0, stream>>>(pfs, 8, idx, r1Wbf, 4096, 256,
                                          r1b, nullptr, 1, VREAL, h256);
    wmma_spiral<<<g256, blk, 0, stream>>>(h256, 8, idx, r1Wbf + 256L * 4096, 4096, 256,
                                          r1b + 256, pfs, 1, VREAL, r1);
    // midDown: fs = (r1 @ W_mid) * zp
    wmma_dense<<<g128, blk, 0, stream>>>(r1, 256, Wmbf, 128,
                                         nullptr, nullptr, 0, VREAL, fs);
    // masked global max pool
    init_gmax<<<1, 256, 0, stream>>>(gmax);
    maxpool_fs<<<dim3((VREAL + 255) / 256, 2), 128, 0, stream>>>(fs, gmax);
    // three 128ch residual spiral blocks (conv2 updates fs in place; rows disjoint per block)
    for (int i = 0; i < 3; ++i) {
        wmma_spiral<<<g128, blk, 0, stream>>>(fs, 7, idx, rsWbf + (2L * i) * 128 * 2048, 2048, 128,
                                              rsb + (2 * i) * 128, nullptr, 1, VREAL, h128);
        wmma_spiral<<<g128, blk, 0, stream>>>(h128, 7, idx, rsWbf + (2L * i + 1) * 128 * 2048, 2048, 128,
                                              rsb + (2 * i + 1) * 128, fs, 1, VREAL, fs);
    }
    // concat [pfs | fs | g]
    concat_h<<<padgrid(2L * VPAD * 512), 256, 0, stream>>>(pfs, fs, gmax, h512);
    // output MLP
    wmma_dense<<<g256, blk, 0, stream>>>(h512, 512, Wo1bf, 256,
                                         bo1, nullptr, 1, 1 << 30, r1);
    wmma_dense<<<g128, blk, 0, stream>>>(r1, 256, Wo2bf, 128,
                                         bo2, nullptr, 1, 1 << 30, h128);
    out_final<<<padgrid(2L * VREAL * 3), 256, 0, stream>>>(h128, Wo3, bo3, (float*)d_out);
}
